// GConvNetFrames_68169720922434
// MI455X (gfx1250) — matrix operand order repaired and checked by reading, not compiled, not run on hardware
//
#include <hip/hip_runtime.h>
#include <hip/hip_bf16.h>
#include <math.h>

#define HID     128
#define TFRAMES 32

typedef __attribute__((ext_vector_type(16))) _Float16 v16h;
typedef __attribute__((ext_vector_type(8)))  float    v8f;

__device__ __forceinline__ float sigmoidf_(float x) {
    // fast: v_exp_f32 + v_rcp_f32
    return __builtin_amdgcn_rcpf(1.0f + __expf(-x));
}
__device__ __forceinline__ float siluf_(float x) { return x * sigmoidf_(x); }

// one-instruction XOR lane swap within 16-lane halves: ds_swizzle_b32 group-of-32 mode
// offset = (xor_mask << 10) | (or_mask << 5) | and_mask
__device__ __forceinline__ float swz_add_x1(float x) {
    return x + __int_as_float(__builtin_amdgcn_ds_swizzle(__float_as_int(x), 0x041f));
}
__device__ __forceinline__ float swz_add_x2(float x) {
    return x + __int_as_float(__builtin_amdgcn_ds_swizzle(__float_as_int(x), 0x081f));
}
__device__ __forceinline__ float swz_add_x4(float x) {
    return x + __int_as_float(__builtin_amdgcn_ds_swizzle(__float_as_int(x), 0x101f));
}
__device__ __forceinline__ float swz_add_x8(float x) {
    return x + __int_as_float(__builtin_amdgcn_ds_swizzle(__float_as_int(x), 0x201f));
}

// ---------------- utility ----------------
__global__ void k_zero(float* p, long n) {
    long i = (long)blockIdx.x * blockDim.x + threadIdx.x;
    if (i < n) p[i] = 0.0f;
}

// ---------------- degrees & norms ----------------
__global__ void k_degree(const int* src, const int* dst, float* degO, float* degI, int E) {
    int e = blockIdx.x * blockDim.x + threadIdx.x;
    if (e < E) {
        atomicAdd(degO + src[e], 1.0f);
        atomicAdd(degI + dst[e], 1.0f);
    }
}
__global__ void k_norm(float* degO, float* degI, int N) {
    int n = blockIdx.x * blockDim.x + threadIdx.x;
    if (n < N) {
        float a = degO[n]; a = a < 1.0f ? 1.0f : a; degO[n] = rsqrtf(a);
        float b = degI[n]; b = b < 1.0f ? 1.0f : b; degI[n] = rsqrtf(b);
    }
}

// ---------------- gconv1 (F=1) ----------------
__global__ void k_edge1(const int* src, const int* dst, const float* f,
                        const float* nO, float* agg1, int E) {
    int e = blockIdx.x * blockDim.x + threadIdx.x;
    if (e < E) {
        int s = src[e];
        atomicAdd(agg1 + dst[e], f[s] * nO[s]);
    }
}

// h32[n][2p..2p+1] = norm_out[n] * silu(agg1[n]*norm_in[n]*W1[j] + b1[j])  packed half2 store
__global__ void k_node1(const float* agg1, const float* nO, const float* nI,
                        const float* W1, const float* b1, unsigned int* h32u, int N) {
    int i = blockIdx.x * blockDim.x + threadIdx.x;
    int n = i >> 4, p = i & 15;
    if (n < N) {
        int j0 = 2 * p, j1 = 2 * p + 1;
        float base = agg1[n] * nI[n];
        float no   = nO[n];
        _Float16 two[2];
        two[0] = (_Float16)(no * siluf_(base * W1[j0] + b1[j0]));
        two[1] = (_Float16)(no * siluf_(base * W1[j1] + b1[j1]));
        h32u[i] = *(const unsigned int*)two;
    }
}

// ---------------- gconv2 edge scatter: packed-f16 atomics ----------------
// One thread per (edge, feature-pair): one b32 gather + one global_atomic_pk_add_f16.
__global__ void k_edge2(const int* __restrict__ src, const int* __restrict__ dst,
                        const unsigned int* __restrict__ h32u,  // [N,16] packed half2
                        unsigned int* __restrict__ agg16u,      // [N,16] packed half2
                        int E) {
    long i = (long)blockIdx.x * blockDim.x + threadIdx.x;
    int e = (int)(i >> 4), p = (int)(i & 15);
    if (e < E) {
        unsigned int v = h32u[(long)src[e] * 16 + p];
        unsigned int* addr = agg16u + (long)dst[e] * 16 + p;
        asm volatile("global_atomic_pk_add_f16 %0, %1, off"
                     :
                     : "v"(addr), "v"(v)
                     : "memory");
    }
}

// ---------------- gconv2 GEMM: [N,32] @ [32,128] via v_wmma_f32_16x16x32_f16 ----------------
// One wave per 16x16 output tile; K=32 is exactly one WMMA.
// A (16-bit 16x32): lanes 0-15 row M=lane, K={0..7,16..23}; lanes 16-31 row M=lane-16, K={8..15,24..31}
// B (16-bit 32x16): lanes 0-15 col N=lane, K={0..7,16..23}; lanes 16-31 col N=lane-16, K={8..15,24..31}
// C/D: elem r: lanes 0-15 -> M=r, N=lane; lanes 16-31 -> M=8+r, N=lane-16
__global__ void __launch_bounds__(32)
k_gemm_wmma(const _Float16* __restrict__ agg16, const float* __restrict__ nI,
            const float* __restrict__ W2, const float* __restrict__ b2,
            const float* __restrict__ gw, float* __restrict__ h128,
            float* __restrict__ gate, int N) {
    int lane = threadIdx.x;
    int hs   = lane >> 4;       // 0: lanes 0-15, 1: lanes 16-31
    int l15  = lane & 15;
    int rowBase = blockIdx.x * 16;
    int m    = rowBase + l15;
    int col  = blockIdx.y * 16 + l15;

    int   mc  = m < N ? m : N - 1;
    float nin = m < N ? nI[mc] : 0.0f;     // zero out-of-range rows via scale
    const _Float16* arow = agg16 + (long)mc * 32;

    v16h a, b;
    int kb0 = hs * 8;
#pragma unroll
    for (int i = 0; i < 8; ++i) a[i]     = (_Float16)((float)arow[kb0 + i] * nin);
#pragma unroll
    for (int i = 0; i < 8; ++i) a[8 + i] = (_Float16)((float)arow[16 + kb0 + i] * nin);

    int kc = hs * 8;
#pragma unroll
    for (int j = 0; j < 8; ++j) {
        int k0 = kc + 2 * j + (j >= 4 ? 8 : 0);
        b[2 * j]     = (_Float16)W2[(k0)     * HID + col];
        b[2 * j + 1] = (_Float16)W2[(k0 + 1) * HID + col];
    }

    v8f c = {};
    c = __builtin_amdgcn_wmma_f32_16x16x32_f16(false, a, false, b, (short)0, c, false, false);

    float bias = b2[col];
    float gwc  = gw[col];

    float vals[8];
#pragma unroll
    for (int r = 0; r < 8; ++r) vals[r] = siluf_(c[r] + bias);

    if (rowBase + 16 <= N) {
        // full tile: straight-line stores, no per-row branches
#pragma unroll
        for (int r = 0; r < 8; ++r)
            h128[(long)(rowBase + hs * 8 + r) * HID + col] = vals[r];
    } else {
        for (int r = 0; r < 8; ++r) {
            int row = rowBase + hs * 8 + r;
            if (row < N) h128[(long)row * HID + col] = vals[r];
        }
    }

    // fused attention-gate dot: reduce val*gw[col] over this wave's 16 columns
#pragma unroll
    for (int r = 0; r < 8; ++r) {
        float g = vals[r] * gwc;
        g = swz_add_x1(g);
        g = swz_add_x2(g);
        g = swz_add_x4(g);
        g = swz_add_x8(g);
        int row = rowBase + hs * 8 + r;
        if (l15 == 0 && row < N) atomicAdd(gate + row, g);
    }
}

// ---------------- softmax reduction (gate bias cancels in softmax) ----------------
__global__ void k_softmax_red(const float* __restrict__ gate, float* __restrict__ red, int N) {
    __shared__ float s[1024];
    int t = threadIdx.x;
    float mx = -1e30f;
    for (int n = t; n < N; n += 1024) mx = fmaxf(mx, gate[n]);
    s[t] = mx; __syncthreads();
    for (int off = 512; off > 0; off >>= 1) {
        if (t < off) s[t] = fmaxf(s[t], s[t + off]);
        __syncthreads();
    }
    float gmax = s[0];
    __syncthreads();
    float sm = 0.0f;
    for (int n = t; n < N; n += 1024) sm += __expf(gate[n] - gmax);
    s[t] = sm; __syncthreads();
    for (int off = 512; off > 0; off >>= 1) {
        if (t < off) s[t] += s[t + off];
        __syncthreads();
    }
    if (t == 0) { red[0] = gmax; red[1] = s[0]; }
}

__global__ void k_pool(const float* __restrict__ h128, const float* __restrict__ gate,
                       const float* __restrict__ red, float* __restrict__ pooled, int N) {
    int c = threadIdx.x;   // 128 threads
    int chunk = (N + gridDim.x - 1) / gridDim.x;
    int n0 = blockIdx.x * chunk;
    int n1 = n0 + chunk; if (n1 > N) n1 = N;
    float gmax = red[0], invZ = __builtin_amdgcn_rcpf(red[1]);
    float acc = 0.0f;
    for (int n = n0; n < n1; ++n) {
        float w = __expf(gate[n] - gmax) * invZ;
        acc += w * h128[(long)n * HID + c];
    }
    atomicAdd(pooled + c, acc);
}

// ---------------- GRU over 32 steps + MLP head ----------------
__global__ void k_gru_head(const float* __restrict__ pooled,
                           const float* __restrict__ W_ih, const float* __restrict__ W_hh,
                           const float* __restrict__ b_ih, const float* __restrict__ b_hh,
                           const float* __restrict__ d1w, const float* __restrict__ d1b,
                           const float* __restrict__ d2w, const float* __restrict__ d2b,
                           float* __restrict__ out) {
    __shared__ float x[128], h[128], o1[16];
    int j = threadIdx.x;
    h[j] = 0.0f;
    __syncthreads();
    for (int t = 0; t < TFRAMES; ++t) {
        x[j] = pooled[t * 128 + j];
        __syncthreads();
        float gr = b_ih[j], gz = b_ih[128 + j], gn = b_ih[256 + j];
        float hr = b_hh[j], hz = b_hh[128 + j], hn = b_hh[256 + j];
        for (int k = 0; k < 128; ++k) {
            float xv = x[k], hv = h[k];
            gr += W_ih[j * 128 + k] * xv;
            gz += W_ih[(128 + j) * 128 + k] * xv;
            gn += W_ih[(256 + j) * 128 + k] * xv;
            hr += W_hh[j * 128 + k] * hv;
            hz += W_hh[(128 + j) * 128 + k] * hv;
            hn += W_hh[(256 + j) * 128 + k] * hv;
        }
        float r  = sigmoidf_(gr + hr);
        float z  = sigmoidf_(gz + hz);
        float nn = tanhf(gn + r * hn);
        float hnew = (1.0f - z) * nn + z * h[j];
        __syncthreads();
        h[j] = hnew;
        __syncthreads();
    }
    if (j < 16) {
        float a = d1b[j];
        for (int k = 0; k < 128; ++k) a += h[k] * d1w[k * 16 + j];
        o1[j] = siluf_(a);
    }
    __syncthreads();
    if (j < 4) {
        float a = d2b[j];
        for (int k = 0; k < 16; ++k) a += o1[k] * d2w[k * 4 + j];
        out[j] = sigmoidf_(siluf_(a));
    }
}

// ---------------- host ----------------
extern "C" void kernel_launch(void* const* d_in, const int* in_sizes, int n_in,
                              void* d_out, int out_size, void* d_ws, size_t ws_size,
                              hipStream_t stream) {
    const float* features = (const float*)d_in[0];
    const int*   src      = (const int*)d_in[1];
    const int*   dst      = (const int*)d_in[2];
    const float* W1  = (const float*)d_in[3];
    const float* b1  = (const float*)d_in[4];
    const float* W2  = (const float*)d_in[5];
    const float* b2  = (const float*)d_in[6];
    const float* gw  = (const float*)d_in[7];
    const float* Wih = (const float*)d_in[9];
    const float* Whh = (const float*)d_in[10];
    const float* bih = (const float*)d_in[11];
    const float* bhh = (const float*)d_in[12];
    const float* d1w = (const float*)d_in[13];
    const float* d1b = (const float*)d_in[14];
    const float* d2w = (const float*)d_in[15];
    const float* d2b = (const float*)d_in[16];

    int N = in_sizes[0] / TFRAMES;
    int E = in_sizes[1];

    // workspace partition (256B aligned)
    char* w = (char*)d_ws;
    auto alloc = [&](size_t bytes) -> char* {
        char* p = w;
        w += (bytes + 255) & ~(size_t)255;
        return p;
    };
    float*        normO  = (float*)alloc((size_t)N * 4);
    float*        normI  = (float*)alloc((size_t)N * 4);
    float*        agg1   = (float*)alloc((size_t)N * 4);
    unsigned int* h32u   = (unsigned int*)alloc((size_t)N * 16 * 4);  // [N,32] f16 packed
    unsigned int* agg16u = (unsigned int*)alloc((size_t)N * 16 * 4);  // [N,32] f16 packed
    float*        h128   = (float*)alloc((size_t)N * HID * 4);
    float*        gate   = (float*)alloc((size_t)N * 4);
    float*        red    = (float*)alloc(256);
    float*        pooled = (float*)alloc((size_t)TFRAMES * HID * 4);

    const int TB = 256;
    auto gsz = [](long n, int tb) { return (unsigned)((n + tb - 1) / tb); };

    // degrees -> norms
    k_zero<<<gsz(N, TB), TB, 0, stream>>>(normO, N);
    k_zero<<<gsz(N, TB), TB, 0, stream>>>(normI, N);
    k_zero<<<gsz(TFRAMES * HID, TB), TB, 0, stream>>>(pooled, TFRAMES * HID);
    k_degree<<<gsz(E, TB), TB, 0, stream>>>(src, dst, normO, normI, E);
    k_norm<<<gsz(N, TB), TB, 0, stream>>>(normO, normI, N);

    dim3 gGemm((N + 15) / 16, HID / 16);

    for (int t = 0; t < TFRAMES; ++t) {
        const float* ft = features + (size_t)t * N;
        k_zero<<<gsz(N, TB), TB, 0, stream>>>(agg1, N);
        k_zero<<<gsz(N, TB), TB, 0, stream>>>(gate, N);
        k_zero<<<gsz((long)N * 16, TB), TB, 0, stream>>>((float*)agg16u, (long)N * 16);
        k_edge1<<<gsz(E, TB), TB, 0, stream>>>(src, dst, ft, normO, agg1, E);
        k_node1<<<gsz((long)N * 16, TB), TB, 0, stream>>>(agg1, normO, normI, W1, b1, h32u, N);
        k_edge2<<<gsz((long)E * 16, TB), TB, 0, stream>>>(src, dst, h32u, agg16u, E);
        k_gemm_wmma<<<gGemm, 32, 0, stream>>>((const _Float16*)agg16u, normI, W2, b2,
                                              gw, h128, gate, N);
        k_softmax_red<<<1, 1024, 0, stream>>>(gate, red, N);
        k_pool<<<32, 128, 0, stream>>>(h128, gate, red, pooled + t * HID, N);
    }

    k_gru_head<<<1, 128, 0, stream>>>(pooled, Wih, Whh, bih, bhh,
                                      d1w, d1b, d2w, d2b, (float*)d_out);
}
